// GroupVectorQuantizer_59828894433281
// MI455X (gfx1250) — matrix operand order repaired and checked by reading, not compiled, not run on hardware
//
#include <hip/hip_runtime.h>

// ---------------- CDNA5 (gfx1250) wave32 WMMA types ----------------
typedef __attribute__((ext_vector_type(16))) _Float16 v16h;
typedef __attribute__((ext_vector_type(8)))  float    v8f;

#define GQ 4          // groups
#define KQ 1024       // codewords per group
#define DQ 64         // codeword dim
#define NQ 32768      // B*T vectors
#define ROWS 128      // latent rows per workgroup
#define CCHUNK 64     // codewords staged per chunk
#define XSTRIDE 68    // f32 LDS row stride (272B: 16B aligned, bank-spread)
#define CSTRIDE 80    // f16 LDS row stride (160B: 32B aligned for v16h)

// Precompute bias[g*K+k] = -0.5 * ||c_k||^2  (argmin dist == argmax (x.c + bias))
__global__ __launch_bounds__(256) void vq_bias(const float* __restrict__ codebooks,
                                               float* __restrict__ bias) {
  int i = blockIdx.x * 256 + threadIdx.x;
  if (i >= GQ * KQ) return;
  const float* c = codebooks + (size_t)i * DQ;
  float s = 0.0f;
#pragma unroll
  for (int j = 0; j < DQ; j += 4) {
    float4 v = *(const float4*)(c + j);
    s += v.x * v.x + v.y * v.y + v.z * v.z + v.w * v.w;
  }
  bias[i] = -0.5f * s;
}

__global__ __launch_bounds__(256) void vq_main(const float* __restrict__ latents,
                                               const float* __restrict__ codebooks,
                                               const float* __restrict__ bias,
                                               float* __restrict__ out,
                                               float* __restrict__ partial) {
  __shared__ float    Xs[ROWS * XSTRIDE];        // 34816 B, exact f32 latent tile
  __shared__ _Float16 Chi[CCHUNK * CSTRIDE];     // 10240 B, codebook hi parts
  __shared__ _Float16 Clo[CCHUNK * CSTRIDE];     // 10240 B, codebook lo parts
  __shared__ float    biasS[CCHUNK];
  __shared__ int      bestIdx[ROWS];
  __shared__ float    wsum[8];

  const int t    = threadIdx.x;
  const int g    = blockIdx.y;
  const int row0 = blockIdx.x * ROWS;
  const int lane = t & 31;
  const int wv   = t >> 5;     // wave 0..7 -> rows [16*wv, 16*wv+16)
  const int nh   = lane >> 4;  // lane half (WMMA K/M selector)
  const int ln   = lane & 15;

  // ---- stage 128x64 latent tile (f32, coalesced float4) ----
#pragma unroll
  for (int i = 0; i < 8; ++i) {
    int q = t + 256 * i;
    int r = q >> 4, c4 = q & 15;
    float4 v = *(const float4*)(latents + (size_t)(row0 + r) * (GQ * DQ) + g * DQ + c4 * 4);
    *(float4*)(Xs + r * XSTRIDE + c4 * 4) = v;
  }
  __syncthreads();

  // ---- build persistent A fragments: x = a_hi + a_lo (fp16 split of fp32) ----
  // 16-bit A 16x32 layout: lane m=ln, half nh; element e -> K = 16*nh+e (contiguous, matches B)
  const int m = wv * 16 + ln;
  const float* xrow = Xs + m * XSTRIDE;
  v16h a0h, a0l, a1h, a1l;
#pragma unroll
  for (int e = 0; e < 16; ++e) {
    int kb = 16 * nh + e;
    float x0 = xrow[kb];
    _Float16 h0 = (_Float16)x0;
    a0h[e] = h0; a0l[e] = (_Float16)(x0 - (float)h0);
    float x1 = xrow[32 + kb];
    _Float16 h1 = (_Float16)x1;
    a1h[e] = h1; a1l[e] = (_Float16)(x1 - (float)h1);
  }

  float vmax[8]; int vidx[8];
#pragma unroll
  for (int r = 0; r < 8; ++r) { vmax[r] = -3.0e38f; vidx[r] = 0; }

  const float* cbg = codebooks + (size_t)g * KQ * DQ;
  const float* bg  = bias + g * KQ;

  for (int chunk = 0; chunk < KQ / CCHUNK; ++chunk) {
    const int k0 = chunk * CCHUNK;
    __syncthreads();  // previous chunk's readers done before overwrite
    // stage 64x64 codebook chunk, converted once to f16 hi/lo
#pragma unroll
    for (int i = 0; i < 4; ++i) {
      int q = t + 256 * i;
      int r = q >> 4, c4 = q & 15;
      float4 v = *(const float4*)(cbg + (size_t)(k0 + r) * DQ + c4 * 4);
      _Float16* dh = Chi + r * CSTRIDE + c4 * 4;
      _Float16* dl = Clo + r * CSTRIDE + c4 * 4;
      float vv[4] = {v.x, v.y, v.z, v.w};
#pragma unroll
      for (int j = 0; j < 4; ++j) {
        _Float16 h = (_Float16)vv[j];
        dh[j] = h;
        dl[j] = (_Float16)(vv[j] - (float)h);
      }
    }
    if (t < CCHUNK) biasS[t] = bg[k0 + t];
    // warm L2->L0 for next chunk (global_prefetch_b8); 256 threads x 64B = 16KB chunk
    if (chunk + 1 < KQ / CCHUNK)
      __builtin_prefetch(cbg + (size_t)(k0 + CCHUNK) * DQ + t * 16, 0, 1);
    __syncthreads();

#pragma unroll
    for (int nt = 0; nt < CCHUNK / 16; ++nt) {
      // B fragment (32x16, f16): lane n=ln; K-run = d0 + 16*nh + [0..15] contiguous
      const _Float16* bh = Chi + (nt * 16 + ln) * CSTRIDE + 16 * nh;
      const _Float16* bl = Clo + (nt * 16 + ln) * CSTRIDE + 16 * nh;
      v16h b0h = *(const v16h*)(bh);
      v16h b0l = *(const v16h*)(bl);
      v16h b1h = *(const v16h*)(bh + 32);
      v16h b1l = *(const v16h*)(bl + 32);
      float bv = biasS[nt * 16 + ln];  // D column N = ln for all 8 acc rows
      v8f acc;
#pragma unroll
      for (int r = 0; r < 8; ++r) acc[r] = bv;
      // x.c in fp32-equivalent precision: hi*hi + hi*lo + lo*hi per 32-deep slab
      acc = __builtin_amdgcn_wmma_f32_16x16x32_f16(false, a0h, false, b0h, (short)0, acc, false, false);
      acc = __builtin_amdgcn_wmma_f32_16x16x32_f16(false, a0h, false, b0l, (short)0, acc, false, false);
      acc = __builtin_amdgcn_wmma_f32_16x16x32_f16(false, a0l, false, b0h, (short)0, acc, false, false);
      acc = __builtin_amdgcn_wmma_f32_16x16x32_f16(false, a1h, false, b1h, (short)0, acc, false, false);
      acc = __builtin_amdgcn_wmma_f32_16x16x32_f16(false, a1h, false, b1l, (short)0, acc, false, false);
      acc = __builtin_amdgcn_wmma_f32_16x16x32_f16(false, a1l, false, b1h, (short)0, acc, false, false);
      const int col = k0 + nt * 16 + ln;  // strictly increasing per lane -> first-min tie-break
#pragma unroll
      for (int r = 0; r < 8; ++r)
        if (acc[r] > vmax[r]) { vmax[r] = acc[r]; vidx[r] = col; }
    }
  }

  // ---- cross-lane argmax within 16-lane halves; D row M = 16*wv + r + 8*nh ----
#pragma unroll
  for (int r = 0; r < 8; ++r) {
    float mv = vmax[r]; int mi = vidx[r];
#pragma unroll
    for (int s = 1; s < 16; s <<= 1) {
      float ov = __shfl_xor(mv, s, 32);
      int   oi = __shfl_xor(mi, s, 32);
      if (ov > mv || (ov == mv && oi < mi)) { mv = ov; mi = oi; }
    }
    if (ln == 0) bestIdx[wv * 16 + r + 8 * nh] = mi;
  }
  __syncthreads();

  // ---- gather winning codewords (forward value of quantized_st) + exact f32 loss ----
  float lsum = 0.0f;
  {
    const int row = t >> 1, hf = t & 1;
    const int best = bestIdx[row];
    const float* cw = cbg + (size_t)best * DQ + hf * 32;
    const float* xr = Xs + row * XSTRIDE + hf * 32;
    float* op = out + (size_t)(row0 + row) * (GQ * DQ) + g * DQ + hf * 32;
#pragma unroll
    for (int j = 0; j < 32; j += 4) {
      float4 c = *(const float4*)(cw + j);
      float4 x = *(const float4*)(xr + j);
      float dx = c.x - x.x, dy = c.y - x.y, dz = c.z - x.z, dw = c.w - x.w;
      lsum += dx * dx + dy * dy + dz * dz + dw * dw;
      *(float4*)(op + j) = c;
    }
  }
#pragma unroll
  for (int s = 1; s < 32; s <<= 1) lsum += __shfl_xor(lsum, s, 32);
  if (lane == 0) wsum[wv] = lsum;
  __syncthreads();
  if (t == 0) {
    float s = 0.0f;
#pragma unroll
    for (int i = 0; i < 8; ++i) s += wsum[i];
    partial[blockIdx.y * gridDim.x + blockIdx.x] = s;  // deterministic 2-pass reduce
  }
}

__global__ __launch_bounds__(256) void vq_finalize(const float* __restrict__ partial,
                                                   float* __restrict__ loss_out) {
  __shared__ float ws[8];
  int t = threadIdx.x;
  float s = 0.0f;
  for (int i = t; i < (NQ / ROWS) * GQ; i += 256) s += partial[i];
#pragma unroll
  for (int sh = 1; sh < 32; sh <<= 1) s += __shfl_xor(s, sh, 32);
  if ((t & 31) == 0) ws[t >> 5] = s;
  __syncthreads();
  if (t == 0) {
    float tot = 0.0f;
#pragma unroll
    for (int i = 0; i < 8; ++i) tot += ws[i];
    // vq_loss = beta*mse + mse = 1.25 * sum / (B*T*G*D)
    *loss_out = tot * (1.25f / 8388608.0f);
  }
}

extern "C" void kernel_launch(void* const* d_in, const int* in_sizes, int n_in,
                              void* d_out, int out_size, void* d_ws, size_t ws_size,
                              hipStream_t stream) {
  const float* latents   = (const float*)d_in[0];  // [8,4096,256] f32
  const float* codebooks = (const float*)d_in[1];  // [4,1024,64]  f32
  float* out     = (float*)d_out;                       // quantized_st flat, then loss
  float* bias    = (float*)d_ws;                        // 4096 f32
  float* partial = (float*)d_ws + GQ * KQ;              // 1024 f32
  float* loss    = out + (size_t)NQ * GQ * DQ;          // element 8388608

  vq_bias<<<dim3((GQ * KQ + 255) / 256), dim3(256), 0, stream>>>(codebooks, bias);
  vq_main<<<dim3(NQ / ROWS, GQ), dim3(256), 0, stream>>>(latents, codebooks, bias, out, partial);
  vq_finalize<<<dim3(1), dim3(256), 0, stream>>>(partial, loss);
}
